// GraphCastProcessorGraphTransformer_62801011802174
// MI455X (gfx1250) — matrix operand order repaired and checked by reading, not compiled, not run on hardware
//
#include <hip/hip_runtime.h>
#include <hip/hip_bf16.h>

// ---------------------------------------------------------------------------
// GraphCast processor transformer for MI455X (gfx1250, wave32, WMMA).
// All GEMMs (QKV/O projections, MLP, QK^T, P*V) use v_wmma_f32_16x16x32_f16.
// Tile staging uses GLOBAL_LOAD_ASYNC_TO_LDS_B128 (ASYNCcnt) when available.
// ---------------------------------------------------------------------------

typedef __attribute__((ext_vector_type(16))) _Float16 v16h;
typedef __attribute__((ext_vector_type(8)))  float    v8f;
typedef __attribute__((ext_vector_type(4)))  int      v4i;

#define NN   4096
#define DD   512
#define NHEAD 4
#define HDIM 128
#define SCALE_F 0.08838834764831845f   // 1/sqrt(128)
#define NEG_F  (-1.0e9f)

union Frag16 { v16h h; uint4 q[2]; };

__device__ __forceinline__ v8f wmma_f16(const v16h a, const v16h b, v8f c) {
  return __builtin_amdgcn_wmma_f32_16x16x32_f16(false, a, false, b, (short)0, c,
                                                false, false);
}

// --- async global->LDS (CDNA5) with sync fallback --------------------------
#if defined(__has_builtin)
#if __has_builtin(__builtin_amdgcn_global_load_async_to_lds_b128)
#define HAVE_ASYNC_LDS 1
#endif
#endif

__device__ __forceinline__ void async_ld128(const void* g, void* l) {
#ifdef HAVE_ASYNC_LDS
  __builtin_amdgcn_global_load_async_to_lds_b128(
      (__attribute__((address_space(1))) v4i*)g,
      (__attribute__((address_space(3))) v4i*)l, 0, 0);
#else
  *(uint4*)l = *(const uint4*)g;
#endif
}

__device__ __forceinline__ void async_wait() {
#ifdef HAVE_ASYNC_LDS
#if __has_builtin(__builtin_amdgcn_s_wait_asynccnt)
  __builtin_amdgcn_s_wait_asynccnt(0);
#else
  asm volatile("s_wait_asynccnt 0" ::: "memory");
#endif
#endif
}

// ---------------------------------------------------------------------------
// f32 -> f16 elementwise convert (weights)
// ---------------------------------------------------------------------------
__global__ __launch_bounds__(256) void convert_f16(const float* __restrict__ in,
                                                   _Float16* __restrict__ out,
                                                   int n) {
  int i = (blockIdx.x * 256 + threadIdx.x) * 4;
  if (i + 3 < n) {
    float4 v = *(const float4*)&in[i];
    out[i + 0] = (_Float16)v.x;
    out[i + 1] = (_Float16)v.y;
    out[i + 2] = (_Float16)v.z;
    out[i + 3] = (_Float16)v.w;
  }
}

// ---------------------------------------------------------------------------
// LayerNorm: one wave (32 lanes) per row of 512, f32 in -> f16 out
// ---------------------------------------------------------------------------
__global__ __launch_bounds__(256) void layernorm_f16(
    const float* __restrict__ x, const float* __restrict__ g,
    const float* __restrict__ b, _Float16* __restrict__ out) {
  int wave = threadIdx.x >> 5;
  int lane = threadIdx.x & 31;
  int row  = blockIdx.x * 8 + wave;
  const float* xr = x + (size_t)row * DD;

  float4 v[4];
  float s = 0.f, sq = 0.f;
#pragma unroll
  for (int i = 0; i < 4; i++) {
    int c = i * 128 + lane * 4;
    v[i] = *(const float4*)&xr[c];
    s  += v[i].x + v[i].y + v[i].z + v[i].w;
    sq += v[i].x * v[i].x + v[i].y * v[i].y + v[i].z * v[i].z + v[i].w * v[i].w;
  }
#pragma unroll
  for (int d = 1; d < 32; d <<= 1) {
    s  += __shfl_xor(s,  d, 32);
    sq += __shfl_xor(sq, d, 32);
  }
  float mu   = s * (1.0f / DD);
  float var  = sq * (1.0f / DD) - mu * mu;
  float rstd = rsqrtf(var + 1e-5f);

  _Float16* orow = out + (size_t)row * DD;
#pragma unroll
  for (int i = 0; i < 4; i++) {
    int c = i * 128 + lane * 4;
    float4 gg = *(const float4*)&g[c];
    float4 bb = *(const float4*)&b[c];
    orow[c + 0] = (_Float16)((v[i].x - mu) * rstd * gg.x + bb.x);
    orow[c + 1] = (_Float16)((v[i].y - mu) * rstd * gg.y + bb.y);
    orow[c + 2] = (_Float16)((v[i].z - mu) * rstd * gg.z + bb.z);
    orow[c + 3] = (_Float16)((v[i].w - mu) * rstd * gg.w + bb.w);
  }
}

// ---------------------------------------------------------------------------
// GEMM: C[4096,512] = act(A16[4096,512] * W16[512,512]^T + bias)
//   ACT:   0 = none, 1 = exact GELU
//   RESID: 0 = write f16 to out16, 1 = out32 += C (f32 residual in-place)
// Block: 256 threads (8 waves), C-tile 128x64; wave tile 32x32 (2x2 WMMA).
// Double-buffered LDS staging via async global->LDS loads.
// ---------------------------------------------------------------------------
template <int ACT, int RESID>
__global__ __launch_bounds__(256) void gemm_nt(
    const _Float16* __restrict__ A, const _Float16* __restrict__ W,
    const float* __restrict__ bias, _Float16* __restrict__ out16,
    float* __restrict__ out32) {
  __shared__ _Float16 As[2][128][72];
  __shared__ _Float16 Bs[2][64][72];

  const int tid  = threadIdx.x;
  const int wave = tid >> 5;
  const int lane = tid & 31;
  const int lm   = lane & 15;
  const int lh   = lane >> 4;
  const int wm   = wave & 3;   // row group (4 waves down)
  const int wn   = wave >> 2;  // col group (2 waves across)
  const int m0   = blockIdx.x * 128;
  const int n0   = blockIdx.y * 64;

  v8f zero = {};
  v8f acc[2][2];
#pragma unroll
  for (int i = 0; i < 2; i++)
#pragma unroll
    for (int j = 0; j < 2; j++) acc[i][j] = zero;

  auto stage = [&](int k0, int bi) {
    // A tile: 128 rows x 64 halfs = 1024 x b128 -> 4 per thread
#pragma unroll
    for (int i = 0; i < 4; i++) {
      int u = tid + 256 * i;
      int r = u >> 3, c = (u & 7) * 8;
      async_ld128(&A[(size_t)(m0 + r) * DD + k0 + c], &As[bi][r][c]);
    }
    // W tile: 64 rows x 64 halfs = 512 x b128 -> 2 per thread
#pragma unroll
    for (int i = 0; i < 2; i++) {
      int u = tid + 256 * i;
      int r = u >> 3, c = (u & 7) * 8;
      async_ld128(&W[(size_t)(n0 + r) * DD + k0 + c], &Bs[bi][r][c]);
    }
  };

  stage(0, 0);
  async_wait();
  __syncthreads();

  for (int s = 0; s < 8; s++) {  // 8 stages of K=64
    const int bi = s & 1;
    if (s < 7) stage((s + 1) * 64, bi ^ 1);  // prefetch next stage

#pragma unroll
    for (int kk = 0; kk < 2; kk++) {
      Frag16 a[2], b[2];
#pragma unroll
      for (int i = 0; i < 2; i++) {
        int row = wm * 32 + 16 * i + lm;
        int kb  = kk * 32 + 8 * lh;
        a[i].q[0] = *(const uint4*)&As[bi][row][kb];
        a[i].q[1] = *(const uint4*)&As[bi][row][kb + 16];
      }
#pragma unroll
      for (int j = 0; j < 2; j++) {
        int nn = wn * 32 + 16 * j + lm;
        int kb = kk * 32 + 8 * lh;
        b[j].q[0] = *(const uint4*)&Bs[bi][nn][kb];
        b[j].q[1] = *(const uint4*)&Bs[bi][nn][kb + 16];
      }
#pragma unroll
      for (int i = 0; i < 2; i++)
#pragma unroll
        for (int j = 0; j < 2; j++)
          acc[i][j] = wmma_f16(a[i].h, b[j].h, acc[i][j]);
    }

    async_wait();
    __syncthreads();
  }

  // epilogue
#pragma unroll
  for (int i = 0; i < 2; i++) {
#pragma unroll
    for (int j = 0; j < 2; j++) {
      int col  = n0 + wn * 32 + 16 * j + lm;
      float bv = bias[col];
#pragma unroll
      for (int r = 0; r < 8; r++) {
        int row = m0 + wm * 32 + 16 * i + r + 8 * lh;
        float v = acc[i][j][r] + bv;
        if (ACT == 1) v = 0.5f * v * (1.0f + erff(v * 0.70710678118654752f));
        if (RESID) {
          size_t idx = (size_t)row * DD + col;
          out32[idx] = out32[idx] + v;
        } else {
          out16[(size_t)row * DD + col] = (_Float16)v;
        }
      }
    }
  }
}

// ---------------------------------------------------------------------------
// Flash attention with arbitrary mask (mask byte != 0 -> score = -1e9).
// Block: one head x 128 query rows; 8 waves, 16 query rows per wave.
// Key tiles of 64, online softmax, all matmuls via WMMA f16.
// K and mask tiles staged via async global->LDS; V transposed manually.
// ---------------------------------------------------------------------------
__global__ __launch_bounds__(256) void flash_attn(
    const _Float16* __restrict__ q16, const _Float16* __restrict__ k16,
    const _Float16* __restrict__ v16, const unsigned char* __restrict__ mask,
    _Float16* __restrict__ o16) {
  __shared__ _Float16 Ks[64][136];        // [key][d]
  __shared__ _Float16 VsT[128][72];       // [d][key] (transposed)
  __shared__ _Float16 Ps[8][16][72];      // per-wave P tile [qrow][key]
  __shared__ unsigned char Ms[128][64];   // mask tile [q][key]

  const int tid  = threadIdx.x;
  const int wave = tid >> 5;
  const int lane = tid & 31;
  const int lm   = lane & 15;
  const int lh   = lane >> 4;
  const int head = blockIdx.y;
  const int q0   = blockIdx.x * 128;
  const int hc   = head * HDIM;
  const int qb   = q0 + wave * 16;

  // Q fragments (A-operand layout), kept in registers for the whole kernel.
  Frag16 qf[4];
#pragma unroll
  for (int kk = 0; kk < 4; kk++) {
    const _Float16* p = &q16[(size_t)(qb + lm) * DD + hc + kk * 32 + 8 * lh];
    qf[kk].q[0] = *(const uint4*)p;
    qf[kk].q[1] = *(const uint4*)(p + 16);
  }

  v8f zero = {};
  v8f of[8];
#pragma unroll
  for (int j = 0; j < 8; j++) of[j] = zero;
  float mrun[8], lrun[8];
#pragma unroll
  for (int r = 0; r < 8; r++) { mrun[r] = -3.0e38f; lrun[r] = 0.f; }

  for (int kt0 = 0; kt0 < NN; kt0 += 64) {
    __syncthreads();
    // K tile: 64 rows x 128 halfs = 1024 x b128 -> 4 per thread (async)
#pragma unroll
    for (int i = 0; i < 4; i++) {
      int u = tid + 256 * i;
      int r = u >> 4, c = (u & 15) * 8;
      async_ld128(&k16[(size_t)(kt0 + r) * DD + hc + c], &Ks[r][c]);
    }
    // mask tile: 128 rows x 64 bytes = 512 x b128 -> 2 per thread (async)
#pragma unroll
    for (int i = 0; i < 2; i++) {
      int u = tid + 256 * i;
      int r = u >> 2, c = (u & 3) * 16;
      async_ld128(&mask[(size_t)(q0 + r) * NN + kt0 + c], &Ms[r][c]);
    }
    // V tile: load rows, store transposed [d][key]
#pragma unroll
    for (int i = 0; i < 4; i++) {
      int u = tid + 256 * i;
      int r = u >> 4, c = (u & 15) * 8;
      union { uint4 u4; _Float16 h[8]; } t;
      t.u4 = *(const uint4*)&v16[(size_t)(kt0 + r) * DD + hc + c];
#pragma unroll
      for (int e = 0; e < 8; e++) VsT[c + e][r] = t.h[e];
    }
    async_wait();
    __syncthreads();

    // S = Q * K^T  (16 q-rows x 64 keys per wave)
    v8f s[4];
#pragma unroll
    for (int j = 0; j < 4; j++) s[j] = zero;
#pragma unroll
    for (int kk = 0; kk < 4; kk++) {
#pragma unroll
      for (int j = 0; j < 4; j++) {
        Frag16 b;
        int kr = 16 * j + lm;
        int kb = kk * 32 + 8 * lh;
        b.q[0] = *(const uint4*)&Ks[kr][kb];
        b.q[1] = *(const uint4*)&Ks[kr][kb + 16];
        s[j] = wmma_f16(qf[kk].h, b.h, s[j]);
      }
    }

    // scale + mask + tile row-max
    float tmax[8];
#pragma unroll
    for (int r = 0; r < 8; r++) tmax[r] = -3.0e38f;
#pragma unroll
    for (int j = 0; j < 4; j++) {
#pragma unroll
      for (int r = 0; r < 8; r++) {
        int ql = wave * 16 + r + 8 * lh;
        int kl = 16 * j + lm;
        float v = s[j][r] * SCALE_F;
        if (Ms[ql][kl]) v = NEG_F;
        s[j][r] = v;
        tmax[r] = fmaxf(tmax[r], v);
      }
    }
#pragma unroll
    for (int d = 1; d < 16; d <<= 1)
#pragma unroll
      for (int r = 0; r < 8; r++)
        tmax[r] = fmaxf(tmax[r], __shfl_xor(tmax[r], d, 32));

    float alpha[8], tsum[8];
#pragma unroll
    for (int r = 0; r < 8; r++) {
      float mn = fmaxf(mrun[r], tmax[r]);
      alpha[r] = __expf(mrun[r] - mn);
      mrun[r]  = mn;
      tsum[r]  = 0.f;
    }
    // P = exp(S - m), write to LDS for A-operand reads
#pragma unroll
    for (int j = 0; j < 4; j++) {
#pragma unroll
      for (int r = 0; r < 8; r++) {
        float p = __expf(s[j][r] - mrun[r]);
        tsum[r] += p;
        Ps[wave][r + 8 * lh][16 * j + lm] = (_Float16)p;
      }
    }
#pragma unroll
    for (int d = 1; d < 16; d <<= 1)
#pragma unroll
      for (int r = 0; r < 8; r++) tsum[r] += __shfl_xor(tsum[r], d, 32);
#pragma unroll
    for (int r = 0; r < 8; r++) lrun[r] = lrun[r] * alpha[r] + tsum[r];
    // rescale running O
#pragma unroll
    for (int f = 0; f < 8; f++)
#pragma unroll
      for (int r = 0; r < 8; r++) of[f][r] *= alpha[r];

    // O += P * V   (A = P [16 x 64], B = V [64 x 128])
#pragma unroll
    for (int kk2 = 0; kk2 < 2; kk2++) {
      Frag16 a;
      int kb = kk2 * 32 + 8 * lh;
      a.q[0] = *(const uint4*)&Ps[wave][lm][kb];
      a.q[1] = *(const uint4*)&Ps[wave][lm][kb + 16];
#pragma unroll
      for (int j2 = 0; j2 < 8; j2++) {
        Frag16 b;
        int dr  = 16 * j2 + lm;
        int kb2 = kk2 * 32 + 8 * lh;
        b.q[0] = *(const uint4*)&VsT[dr][kb2];
        b.q[1] = *(const uint4*)&VsT[dr][kb2 + 16];
        of[j2] = wmma_f16(a.h, b.h, of[j2]);
      }
    }
  }

  float inv[8];
#pragma unroll
  for (int r = 0; r < 8; r++) inv[r] = 1.0f / lrun[r];
#pragma unroll
  for (int j2 = 0; j2 < 8; j2++)
#pragma unroll
    for (int r = 0; r < 8; r++)
      o16[(size_t)(qb + r + 8 * lh) * DD + hc + 16 * j2 + lm] =
          (_Float16)(of[j2][r] * inv[r]);
}

// ---------------------------------------------------------------------------
// Host orchestration
// ---------------------------------------------------------------------------
extern "C" void kernel_launch(void* const* d_in, const int* in_sizes, int n_in,
                              void* d_out, int out_size, void* d_ws,
                              size_t ws_size, hipStream_t stream) {
  (void)in_sizes; (void)n_in; (void)out_size; (void)ws_size;

  const float* nfeat = (const float*)d_in[0];
  const unsigned char* maskp = (const unsigned char*)d_in[1];
  const float* ln1_g = (const float*)d_in[2];
  const float* ln1_b = (const float*)d_in[3];
  const float* wq = (const float*)d_in[4];
  const float* bq = (const float*)d_in[5];
  const float* wk = (const float*)d_in[6];
  const float* bk = (const float*)d_in[7];
  const float* wv = (const float*)d_in[8];
  const float* bv = (const float*)d_in[9];
  const float* wo = (const float*)d_in[10];
  const float* bo = (const float*)d_in[11];
  const float* ln2_g = (const float*)d_in[12];
  const float* ln2_b = (const float*)d_in[13];
  const float* fc1_w = (const float*)d_in[14];
  const float* fc1_b = (const float*)d_in[15];
  const float* fc2_w = (const float*)d_in[16];
  const float* fc2_b = (const float*)d_in[17];

  float* x = (float*)d_out;  // residual stream lives in d_out

  // workspace carve-up
  const size_t WELEMS = 2 * 512 * 512;      // per weight tensor (L=2)
  const size_t AELEMS = (size_t)NN * DD;    // activation tensor
  char* w = (char*)d_ws;
  _Float16* wq16  = (_Float16*)(w);                    // 1 MiB each
  _Float16* wk16  = (_Float16*)(w + 1  * 1048576);
  _Float16* wv16  = (_Float16*)(w + 2  * 1048576);
  _Float16* wo16  = (_Float16*)(w + 3  * 1048576);
  _Float16* w116  = (_Float16*)(w + 4  * 1048576);
  _Float16* w216  = (_Float16*)(w + 5  * 1048576);
  _Float16* h16   = (_Float16*)(w + 6  * 1048576);     // 4 MiB each
  _Float16* q16b  = (_Float16*)(w + 10 * 1048576);
  _Float16* k16b  = (_Float16*)(w + 14 * 1048576);
  _Float16* v16b  = (_Float16*)(w + 18 * 1048576);
  _Float16* o16b  = (_Float16*)(w + 22 * 1048576);
  _Float16* m16b  = (_Float16*)(w + 26 * 1048576);

  // x <- nfeat
  (void)hipMemcpyAsync(x, nfeat, AELEMS * sizeof(float),
                       hipMemcpyDeviceToDevice, stream);

  // convert all weights to f16 once
  const int cgrid = (int)(WELEMS / (256 * 4));
  convert_f16<<<cgrid, 256, 0, stream>>>(wq, wq16, (int)WELEMS);
  convert_f16<<<cgrid, 256, 0, stream>>>(wk, wk16, (int)WELEMS);
  convert_f16<<<cgrid, 256, 0, stream>>>(wv, wv16, (int)WELEMS);
  convert_f16<<<cgrid, 256, 0, stream>>>(wo, wo16, (int)WELEMS);
  convert_f16<<<cgrid, 256, 0, stream>>>(fc1_w, w116, (int)WELEMS);
  convert_f16<<<cgrid, 256, 0, stream>>>(fc2_w, w216, (int)WELEMS);

  const dim3 ggrid(NN / 128, DD / 64);  // 32 x 8
  const dim3 agrid(NN / 128, NHEAD);    // 32 x 4

  for (int l = 0; l < 2; l++) {
    const size_t wl = (size_t)l * 512 * 512;
    const size_t bl = (size_t)l * 512;

    // pre-LN + attention
    layernorm_f16<<<NN / 8, 256, 0, stream>>>(x, ln1_g + bl, ln1_b + bl, h16);
    gemm_nt<0, 0><<<ggrid, 256, 0, stream>>>(h16, wq16 + wl, bq + bl, q16b,
                                             nullptr);
    gemm_nt<0, 0><<<ggrid, 256, 0, stream>>>(h16, wk16 + wl, bk + bl, k16b,
                                             nullptr);
    gemm_nt<0, 0><<<ggrid, 256, 0, stream>>>(h16, wv16 + wl, bv + bl, v16b,
                                             nullptr);
    flash_attn<<<agrid, 256, 0, stream>>>(q16b, k16b, v16b, maskp, o16b);
    gemm_nt<0, 1><<<ggrid, 256, 0, stream>>>(o16b, wo16 + wl, bo + bl, nullptr,
                                             x);

    // pre-LN + MLP
    layernorm_f16<<<NN / 8, 256, 0, stream>>>(x, ln2_g + bl, ln2_b + bl, h16);
    gemm_nt<1, 0><<<ggrid, 256, 0, stream>>>(h16, w116 + wl, fc1_b + bl, m16b,
                                             nullptr);
    gemm_nt<0, 1><<<ggrid, 256, 0, stream>>>(m16b, w216 + wl, fc2_b + bl,
                                             nullptr, x);
  }
}
